// MultiHeadAttentionBlock_18107582120455
// MI455X (gfx1250) — matrix operand order repaired and checked by reading, not compiled, not run on hardware
//
#include <hip/hip_runtime.h>
#include <hip/hip_bf16.h>

// Problem constants (match reference)
#define B_  2
#define S_  2048
#define D_  1024
#define H_  16
#define DK_ 64

typedef __attribute__((ext_vector_type(16))) __bf16 v16bf;
typedef __attribute__((ext_vector_type(8)))  float  v8f;
typedef __attribute__((ext_vector_type(4)))  unsigned int u32x4;
typedef __attribute__((ext_vector_type(8)))  int    i32x8;
typedef __attribute__((ext_vector_type(4)))  int    i32x4;

#if __has_builtin(__builtin_amdgcn_tensor_load_to_lds)
#define HAVE_TDM 1
#if __has_include(<hip/amd_detail/amd_gfx1250_TDM.h>)
#define TDM_ARGS6 1
#endif
#endif

static __device__ __forceinline__ v8f wmma_bf16(v16bf a, v16bf b, v8f c) {
  // 8 args: (neg_a, A, neg_b, B, c_mod, C, reuse_a, reuse_b)
  return __builtin_amdgcn_wmma_f32_16x16x32_bf16(false, a, false, b, (short)0, c,
                                                 false, false);
}

static __device__ __forceinline__ void wait_tensorcnt0() {
#if __has_builtin(__builtin_amdgcn_s_wait_tensorcnt)
  __builtin_amdgcn_s_wait_tensorcnt(0);
#else
  asm volatile("s_wait_tensorcnt 0x0" ::: "memory");
#endif
}

#ifdef HAVE_TDM
// TDM: DMA a 64x64 bf16 tile (contiguous rows of 64 elements, row stride 64)
// from global memory into LDS, padding each 128B row with 32B so LDS rows land
// on an 80-element (160B) stride. D# layout per CDNA5 ISA.
static __device__ __forceinline__ void tdm_load_tile64x64_bf16(
    unsigned long long gaddr, unsigned lds_addr)
{
  u32x4 g0;
  g0[0] = 1u;                                            // count=1 (valid), user mode
  g0[1] = lds_addr;                                      // lds_addr [63:32]
  g0[2] = (unsigned)(gaddr & 0xFFFFFFFFull);             // global_addr [95:64]
  g0[3] = (unsigned)((gaddr >> 32) & 0x01FFFFFFull)      // global_addr [120:96]
        | 0x80000000u;                                   // type=2 ("image") [127:126]

  i32x8 g1;
  // [17:16] data_size=1 (2B)  [20] pad_enable=1
  // [24:22] pad_interval=4 (32 DWORDs = 128B)  [31:25] pad_amount=7 (8 DWORDs = 32B)
  g1[0] = (1 << 16) | (1 << 20) | (4 << 22) | (7 << 25);
  g1[1] = (int)(64u << 16);   // [47:32] atomic_barrier_addr=0 ; [63:48] tensor_dim0.lo = 64
  g1[2] = (int)(64u << 16);   // [79:64] tensor_dim0.hi=0 ; [95:80] tensor_dim1.lo = 64
  g1[3] = (int)(64u << 16);   // [111:96] tensor_dim1.hi=0 ; [127:112] tile_dim0 = 64
  g1[4] = 64;                 // [143:128] tile_dim1 = 64 ; [159:144] tile_dim2 = 0
  g1[5] = 64;                 // [191:160] tensor_dim0_stride.lo = 64
  g1[6] = 0;                  // [207:192] stride.hi=0 ; [223:208] tensor_dim1_stride.lo=0
  g1[7] = 0;

  i32x4 gz = {0, 0, 0, 0};    // groups 2/3 unused (2D tensor)
#ifdef TDM_ARGS6
  i32x8 gz8 = {0, 0, 0, 0, 0, 0, 0, 0};
  __builtin_amdgcn_tensor_load_to_lds(g0, g1, gz, gz, gz8, 0);
#else
  __builtin_amdgcn_tensor_load_to_lds(g0, g1, gz, gz, 0);
#endif
}
#endif // HAVE_TDM

// ---------------------------------------------------------------------------
// GEMM: C[M,N] = A[M,K] @ W[N,K]^T + bias
// mode 0: store fp32 [M,N] to outF
// mode 1: store bf16 head-split [B,H,S,DK] to outH
// Block: 256 threads (8 waves). Block tile 128x128, wave tile 32x64,
// K staged in 32-wide bf16 LDS tiles. 8 v_wmma per wave per K-step.
// (fp32->bf16 conversion forces the VGPR staging path here; TDM is used in
//  the attention kernel where staging is a pure bf16 copy.)
// ---------------------------------------------------------------------------
__global__ __launch_bounds__(256) void gemm_xwt_kernel(
    const float* __restrict__ A,
    const float* __restrict__ W,
    const float* __restrict__ bias,
    float* __restrict__ outF,
    __bf16* __restrict__ outH,
    int M, int N, int K, int mode)
{
  __shared__ __bf16 sA[128][40];   // row stride 80B (16B aligned)
  __shared__ __bf16 sB[128][40];

  const int tid  = threadIdx.x;
  const int lane = tid & 31;
  const int wave = tid >> 5;        // 0..7
  const bool hi  = lane >= 16;
  const int l16  = lane & 15;
  const int wm   = wave >> 1;       // 0..3  -> 32-row strip
  const int wn   = wave & 1;        // 0..1  -> 64-col strip

  const int m0 = blockIdx.x * 128;
  const int n0 = blockIdx.y * 128;

  const int r  = tid >> 1;          // 0..127 staging row
  const int kh = (tid & 1) * 16;    // staging k-half: 0 or 16

  v8f zero = {0.f,0.f,0.f,0.f,0.f,0.f,0.f,0.f};
  v8f acc[2][4];
#pragma unroll
  for (int mi = 0; mi < 2; ++mi)
#pragma unroll
    for (int ni = 0; ni < 4; ++ni)
      acc[mi][ni] = zero;

  for (int k0 = 0; k0 < K; k0 += 32) {
    // ---- stage A tile [128x32] and W tile [128x32] as bf16 in LDS ----
    {
      const float4* a4 = (const float4*)(A + (size_t)(m0 + r) * K + k0 + kh);
      const float4* w4 = (const float4*)(W + (size_t)(n0 + r) * K + k0 + kh);
#pragma unroll
      for (int c = 0; c < 4; ++c) {
        float4 fa = a4[c];
        float4 fw = w4[c];
        sA[r][kh + c*4 + 0] = (__bf16)fa.x;
        sA[r][kh + c*4 + 1] = (__bf16)fa.y;
        sA[r][kh + c*4 + 2] = (__bf16)fa.z;
        sA[r][kh + c*4 + 3] = (__bf16)fa.w;
        sB[r][kh + c*4 + 0] = (__bf16)fw.x;
        sB[r][kh + c*4 + 1] = (__bf16)fw.y;
        sB[r][kh + c*4 + 2] = (__bf16)fw.z;
        sB[r][kh + c*4 + 3] = (__bf16)fw.w;
      }
      if (k0 + 64 < K) {
        __builtin_prefetch(A + (size_t)(m0 + r) * K + k0 + 64, 0, 1);
        __builtin_prefetch(W + (size_t)(n0 + r) * K + k0 + 64, 0, 1);
      }
    }
    __syncthreads();

    // ---- gather fragments per ISA VGPR layouts ----
    v16bf af[2], bfr[4];
#pragma unroll
    for (int mi = 0; mi < 2; ++mi) {
      const int row = wm*32 + mi*16 + l16;
      const int kb  = hi ? 8 : 0;         // lanes>=16 hold K{8..15,24..31}
#pragma unroll
      for (int t = 0; t < 8; ++t) {
        af[mi][t]     = sA[row][kb + t];
        af[mi][8 + t] = sA[row][kb + t + 16];
      }
    }
#pragma unroll
    for (int ni = 0; ni < 4; ++ni) {
      const int col = wn*64 + ni*16 + l16;
      const int kb  = hi ? 8 : 0;         // lanes>=16 hold K{8..15,24..31}
#pragma unroll
      for (int t = 0; t < 16; ++t)
        bfr[ni][t] = sB[col][kb + (t < 8 ? t : t + 8)];
    }

#pragma unroll
    for (int mi = 0; mi < 2; ++mi)
#pragma unroll
      for (int ni = 0; ni < 4; ++ni)
        acc[mi][ni] = wmma_bf16(af[mi], bfr[ni], acc[mi][ni]);

    __syncthreads();
  }

  // ---- epilogue: C layout (VGPR r -> row r / r+8 by lane half, col = l16) ----
#pragma unroll
  for (int mi = 0; mi < 2; ++mi) {
#pragma unroll
    for (int ni = 0; ni < 4; ++ni) {
      const int n = n0 + wn*64 + ni*16 + l16;
      const float bv = bias[n];
#pragma unroll
      for (int rr = 0; rr < 8; ++rr) {
        const int m = m0 + wm*32 + mi*16 + rr + (hi ? 8 : 0);
        const float val = acc[mi][ni][rr] + bv;
        if (mode == 0) {
          outF[(size_t)m * N + n] = val;
        } else {
          const int bb = m / S_, ss = m % S_;
          const int hh = n >> 6, dk = n & 63;
          outH[(((size_t)bb * H_ + hh) * S_ + ss) * DK_ + dk] = (__bf16)val;
        }
      }
    }
  }
}

// ---------------------------------------------------------------------------
// Flash attention: one block per (b*H+h, 64 Q rows). 128 threads = 4 waves,
// each wave owns a 16-row Q strip. Streams 64-key tiles of K/V through LDS
// via the Tensor Data Mover (falls back to vector copies if TDM builtin is
// unavailable), online softmax, WMMA for both QK^T and PV.
// ---------------------------------------------------------------------------
__global__ __launch_bounds__(128) void flash_attn_kernel(
    const __bf16* __restrict__ Qb,   // [B,H,S,DK] bf16
    const __bf16* __restrict__ Kb,
    const __bf16* __restrict__ Vb,
    const int*    __restrict__ mask, // [B,1,S,S]
    float*        __restrict__ O)    // [B,S,D] fp32
{
  __shared__ __bf16 sK[64][DK_ + 16];       // row stride 160B, 16B aligned
  __shared__ __bf16 sV[64][DK_ + 16];
  __shared__ __bf16 sP[4][16][64 + 16];     // per-wave P staging

  const int tid  = threadIdx.x;
  const int lane = tid & 31;
  const int wave = tid >> 5;        // 0..3
  const bool hi  = lane >= 16;
  const int l16  = lane & 15;

  const int bh = blockIdx.y;        // 0..B*H-1
  const int b  = bh / H_;
  const int h  = bh % H_;
  const int q0 = blockIdx.x * 64 + wave * 16;

  // ---- load Q A-fragments for the wave's 16 rows (2 K-chunks over DK=64) ----
  v16bf aq[2];
#pragma unroll
  for (int kc = 0; kc < 2; ++kc) {
    const int row = q0 + l16;
    const __bf16* qrow = Qb + (((size_t)bh * S_) + row) * DK_ + kc * 32;
    const int kb = hi ? 8 : 0;
#pragma unroll
    for (int t = 0; t < 8; ++t) {
      aq[kc][t]     = qrow[kb + t];
      aq[kc][8 + t] = qrow[kb + t + 16];
    }
  }

  v8f zero = {0.f,0.f,0.f,0.f,0.f,0.f,0.f,0.f};
  v8f o_acc[4];
#pragma unroll
  for (int ni = 0; ni < 4; ++ni) o_acc[ni] = zero;

  float mrow[8], lrow[8];
#pragma unroll
  for (int rr = 0; rr < 8; ++rr) { mrow[rr] = -3.0e38f; lrow[rr] = 0.f; }

  const int* mb = mask + (size_t)b * S_ * S_;

  for (int kt = 0; kt < S_ / 64; ++kt) {
    // ---- stage K,V tile [64 keys x 64 d] into LDS ----
#ifdef HAVE_TDM
    if (wave == 0) {
      // Descriptor values are wave-uniform (kernarg base + blockIdx + kt).
      const unsigned long long gK =
          (unsigned long long)(size_t)(Kb + (((size_t)bh * S_) + kt * 64) * DK_);
      const unsigned long long gV =
          (unsigned long long)(size_t)(Vb + (((size_t)bh * S_) + kt * 64) * DK_);
      tdm_load_tile64x64_bf16(gK, (unsigned)(size_t)&sK[0][0]);
      tdm_load_tile64x64_bf16(gV, (unsigned)(size_t)&sV[0][0]);
      wait_tensorcnt0();
    }
    __syncthreads();
#else
    {
      const int key0 = kt * 64;
      const int rr = tid >> 1;          // 0..63
      const int dh = (tid & 1) * 32;    // 0 / 32
      const uint4* ks = (const uint4*)(Kb + (((size_t)bh * S_) + key0 + rr) * DK_ + dh);
      const uint4* vs = (const uint4*)(Vb + (((size_t)bh * S_) + key0 + rr) * DK_ + dh);
      uint4* dk4 = (uint4*)&sK[rr][dh];
      uint4* dv4 = (uint4*)&sV[rr][dh];
#pragma unroll
      for (int c = 0; c < 4; ++c) { dk4[c] = ks[c]; dv4[c] = vs[c]; }
    }
    __syncthreads();
#endif

    // ---- S = Q K^T : 4 key-subtiles of 16, 2 WMMA each (DK=64) ----
    v8f sc[4];
#pragma unroll
    for (int ct = 0; ct < 4; ++ct) sc[ct] = zero;
#pragma unroll
    for (int ct = 0; ct < 4; ++ct) {
#pragma unroll
      for (int kc = 0; kc < 2; ++kc) {
        v16bf bk;
        const int key = ct*16 + l16;
        const int kb  = kc*32 + (hi ? 8 : 0);
#pragma unroll
        for (int t = 0; t < 16; ++t) bk[t] = sK[key][kb + (t < 8 ? t : t + 8)];
        sc[ct] = wmma_bf16(aq[kc], bk, sc[ct]);
      }
    }

    // ---- mask + scale + online softmax over this 64-key tile ----
    float p[4][8];
#pragma unroll
    for (int rr = 0; rr < 8; ++rr) {
      const int sq = q0 + rr + (hi ? 8 : 0);
      float mx = -3.0e38f;
#pragma unroll
      for (int ct = 0; ct < 4; ++ct) {
        const int sk = kt*64 + ct*16 + l16;
        float v = sc[ct][rr] * 0.125f;               // 1/sqrt(64)
        if (mb[(size_t)sq * S_ + sk] == 0) v = -1.0e9f;
        p[ct][rr] = v;
        mx = fmaxf(mx, v);
      }
#pragma unroll
      for (int off = 1; off < 16; off <<= 1)
        mx = fmaxf(mx, __shfl_xor(mx, off, 32));
      const float mnew  = fmaxf(mrow[rr], mx);
      const float alpha = __expf(mrow[rr] - mnew);
      mrow[rr] = mnew;
      float sum = 0.f;
#pragma unroll
      for (int ct = 0; ct < 4; ++ct) {
        const float e = __expf(p[ct][rr] - mnew);
        p[ct][rr] = e;
        sum += e;
      }
#pragma unroll
      for (int off = 1; off < 16; off <<= 1)
        sum += __shfl_xor(sum, off, 32);
      lrow[rr] = lrow[rr] * alpha + sum;
#pragma unroll
      for (int ni = 0; ni < 4; ++ni) o_acc[ni][rr] *= alpha;
      // park P in LDS (C-layout position) for A-layout re-gather
#pragma unroll
      for (int ct = 0; ct < 4; ++ct)
        sP[wave][rr + (hi ? 8 : 0)][ct*16 + l16] = (__bf16)p[ct][rr];
    }

    // ---- O += P @ V : contraction over 64 keys (2 chunks), DK in 4 subtiles ----
#pragma unroll
    for (int kc = 0; kc < 2; ++kc) {
      v16bf ap;
      const int kb0 = kc*32 + (hi ? 8 : 0);
#pragma unroll
      for (int t = 0; t < 8; ++t) {
        ap[t]     = sP[wave][l16][kb0 + t];
        ap[8 + t] = sP[wave][l16][kb0 + t + 16];
      }
#pragma unroll
      for (int ni = 0; ni < 4; ++ni) {
        v16bf bv;
        const int n  = ni*16 + l16;
        const int kb = kc*32 + (hi ? 8 : 0);
#pragma unroll
        for (int t = 0; t < 16; ++t) bv[t] = sV[kb + (t < 8 ? t : t + 8)][n];
        o_acc[ni] = wmma_bf16(ap, bv, o_acc[ni]);
      }
    }
    __syncthreads();
  }

  // ---- epilogue: normalize and write fp32 [B,S,D] ----
#pragma unroll
  for (int ni = 0; ni < 4; ++ni) {
#pragma unroll
    for (int rr = 0; rr < 8; ++rr) {
      const int sq   = q0 + rr + (hi ? 8 : 0);
      const int dcol = h * DK_ + ni*16 + l16;
      const float v  = o_acc[ni][rr] / fmaxf(lrow[rr], 1e-30f);
      O[((size_t)b * S_ + sq) * D_ + dcol] = v;
    }
  }
}

// ---------------------------------------------------------------------------
extern "C" void kernel_launch(void* const* d_in, const int* in_sizes, int n_in,
                              void* d_out, int out_size, void* d_ws, size_t ws_size,
                              hipStream_t stream)
{
  (void)in_sizes; (void)n_in; (void)out_size; (void)ws_size;

  const float* q    = (const float*)d_in[0];
  const float* k    = (const float*)d_in[1];
  const float* v    = (const float*)d_in[2];
  const int*   mask = (const int*)  d_in[3];
  const float* wq   = (const float*)d_in[4];
  const float* bq   = (const float*)d_in[5];
  const float* wk   = (const float*)d_in[6];
  const float* bk   = (const float*)d_in[7];
  const float* wv   = (const float*)d_in[8];
  const float* bv   = (const float*)d_in[9];
  const float* wo   = (const float*)d_in[10];
  const float* bo   = (const float*)d_in[11];
  float* out = (float*)d_out;

  // Workspace layout: Qb,Kb,Vb bf16 head-split (8MB each) + Of fp32 (16MB)
  const size_t nper = (size_t)B_ * H_ * S_ * DK_;   // 4M elements
  __bf16* Qb = (__bf16*)d_ws;
  __bf16* Kb = Qb + nper;
  __bf16* Vb = Kb + nper;
  float*  Of = (float*)(Vb + nper);

  const int M = B_ * S_;   // 4096
  const int N = D_;        // 1024
  const int K = D_;        // 1024
  dim3 gg(M / 128, N / 128);   // (32, 8)

  gemm_xwt_kernel<<<gg, 256, 0, stream>>>(q, wq, bq, nullptr, Qb, M, N, K, 1);
  gemm_xwt_kernel<<<gg, 256, 0, stream>>>(k, wk, bk, nullptr, Kb, M, N, K, 1);
  gemm_xwt_kernel<<<gg, 256, 0, stream>>>(v, wv, bv, nullptr, Vb, M, N, K, 1);

  flash_attn_kernel<<<dim3(S_ / 64, B_ * H_), 128, 0, stream>>>(Qb, Kb, Vb, mask, Of);

  gemm_xwt_kernel<<<gg, 256, 0, stream>>>(Of, wo, bo, out, nullptr, M, N, K, 0);
}
